// MultiHeadAttention_8821862826478
// MI455X (gfx1250) — matrix operand order repaired and checked by reading, not compiled, not run on hardware
//
#include <hip/hip_runtime.h>
#include <hip/hip_bf16.h>

typedef __bf16 bf16;
typedef __attribute__((ext_vector_type(16))) __bf16 v16bf;
typedef __attribute__((ext_vector_type(8)))  __bf16 v8bf;
typedef __attribute__((ext_vector_type(8)))  float   v8f;

__device__ __forceinline__ v8f wmma_bf16(v16bf a, v16bf b, v8f c) {
  // D = A(16x32 bf16) x B(32x16 bf16) + C(16x16 f32)
  return __builtin_amdgcn_wmma_f32_16x16x32_bf16(
      /*neg_a=*/false, a, /*neg_b=*/false, b,
      /*c_mod=*/(short)0, c, /*reuse_a=*/false, /*reuse_b=*/false);
}

__device__ __forceinline__ v16bf cat16(v8bf lo, v8bf hi) {
  return __builtin_shufflevector(lo, hi, 0, 1, 2, 3, 4, 5, 6, 7,
                                         8, 9, 10, 11, 12, 13, 14, 15);
}

// A operand from row-major [M][K] storage (two ds_load_b128).
// Lanes 0-15: M=lane, elems 0..7 -> K=k0+0..7, elems 8..15 -> K=k0+16..23;
// lanes 16-31: +8 on K.
__device__ __forceinline__ v16bf fragA(const bf16* m, int ld, int row, int k0, int lane) {
  const int off = (lane & 16) ? 8 : 0;
  const bf16* p = m + row * ld + k0 + off;
  return cat16(*(const v8bf*)p, *(const v8bf*)(p + 16));
}

// B operand from row-major [N][K] storage, i.e. B = stored^T (two ds_load_b128).
// Lane holds N=lane%16; elems 0..7 -> K = k0 + off + 0..7, elems 8..15 -> K = k0 + off + 16..23, off = 8 for lanes 16-31.
__device__ __forceinline__ v16bf fragB(const bf16* m, int ld, int nrow, int k0, int lane) {
  const int off = (lane & 16) ? 8 : 0;
  const bf16* p = m + nrow * ld + k0 + off;
  return cat16(*(const v8bf*)p, *(const v8bf*)(p + 16));
}

#define BDIM 256

__global__ __launch_bounds__(256) void mha_w_attn_kernel(
    const float* __restrict__ qg, const float* __restrict__ kg, const float* __restrict__ vg,
    const float* __restrict__ wq, const float* __restrict__ bq,
    const float* __restrict__ wk, const float* __restrict__ bk,
    const float* __restrict__ wv, const float* __restrict__ bv,
    float* __restrict__ outg)
{
  constexpr int C = 64, W = 128;
  constexpr int HW = 128 * 128;
  constexpr int LDR = 72;   // sRawT [W][LDR]  bf16, input slice transposed (pixel-major)
  constexpr int LDW = 72;   // sW    [C][LDW]  bf16, weights [o][c] (native)
  constexpr int LDP = 72;   // sQp/sKp [W][LDP] bf16, projections pixel-major
  constexpr int LDV = 136;  // sVpT  [C][LDV]  bf16, v-projection channel-major
  constexpr int LDA = 136;  // sAttn [W][LDA]  bf16
  constexpr int LDO = 68;   // sOut  [W][LDO]  f32

  __shared__ bf16  sRawT[W * LDR];
  __shared__ bf16  sW[C * LDW];
  __shared__ float sBias[C];
  __shared__ bf16  sQp[W * LDP];
  __shared__ bf16  sKp[W * LDP];
  __shared__ bf16  sVpT[C * LDV];
  __shared__ bf16  sAttn[W * LDA];
  __shared__ float sOut[W * LDO];

  const int tid   = threadIdx.x;
  const int lane  = tid & 31;
  const int wid   = tid >> 5;
  const int r16   = lane & 15;
  const int halfA = (lane & 16) ? 8 : 0;   // D-row (M) offset for upper lane half

  const int bh = blockIdx.x;               // b*128 + h
  const size_t base = (size_t)(bh >> 7) * (size_t)(C * HW) + (size_t)(bh & 127) * W;

  const float* Xg[3] = { qg, kg, vg };
  const float* Wg[3] = { wq, wk, wv };
  const float* Bg[3] = { bq, bk, bv };

  // ---------------- Stage 1: three projections ----------------
  // D[w][o] = sum_c X^T[w][c] * W[o][c]
  for (int p = 0; p < 3; ++p) {
    if (p) __syncthreads();               // protect sRawT/sW reuse
    // stage weights 64x64 f32 -> bf16, native [o][c] layout
    for (int t = tid; t < C * 16; t += BDIM) {
      int o = t >> 4, c4 = (t & 15) << 2;
      const float4 f = *(const float4*)(Wg[p] + o * C + c4);
      bf16* d = &sW[o * LDW + c4];
      d[0] = (bf16)f.x; d[1] = (bf16)f.y; d[2] = (bf16)f.z; d[3] = (bf16)f.w;
    }
    if (tid < C) sBias[tid] = Bg[p][tid];
    // stage raw [c][w] f32 -> transposed LDS [w][c] bf16 (coalesced global reads)
    for (int t = tid; t < C * (W / 4); t += BDIM) {
      int c = t >> 5, w = (t & 31) << 2;
      const float4 f = *(const float4*)(Xg[p] + base + (size_t)c * HW + w);
      sRawT[(w + 0) * LDR + c] = (bf16)f.x;
      sRawT[(w + 1) * LDR + c] = (bf16)f.y;
      sRawT[(w + 2) * LDR + c] = (bf16)f.z;
      sRawT[(w + 3) * LDR + c] = (bf16)f.w;
    }
    __syncthreads();

    bf16* Ppix = p ? sKp : sQp;           // runtime select, no static addrspacecast
#pragma unroll
    for (int ti = 0; ti < 4; ++ti) {
      int tile = wid + 8 * ti;            // 0..31 : 8 w-tiles (M) x 4 o-tiles (N)
      int mw = tile & 7, no = tile >> 3;
      v8f acc = {};
#pragma unroll
      for (int k0 = 0; k0 < C; k0 += 32) {
        v16bf a = fragA(sRawT, LDR, mw * 16 + r16, k0, lane);  // A = X^T [m=w][k=c]
        v16bf b = fragB(sW, LDW, no * 16 + r16, k0, lane);     // B = W^T from [n=o][k=c]
        acc = wmma_bf16(a, b, acc);
      }
      const float bb = sBias[no * 16 + r16];                   // n = o fixed per lane
      if (p == 2) {
        // vp stored channel-major [c][x]: contiguous 16B store (m = x across j)
        v8bf pk;
#pragma unroll
        for (int j = 0; j < 8; ++j) pk[j] = (bf16)(acc[j] + bb);
        *(v8bf*)&sVpT[(no * 16 + r16) * LDV + mw * 16 + halfA] = pk;
      } else {
        // qp/kp stored pixel-major [w][c]: transpose store (row = m varies)
#pragma unroll
        for (int j = 0; j < 8; ++j)
          Ppix[(mw * 16 + halfA + j) * LDP + no * 16 + r16] = (bf16)(acc[j] + bb);
      }
    }
  }
  __syncthreads();

  // ---------------- Stage 2: attn[w][x] = sum_c qp[w][c]*kp[x][c] ----------------
#pragma unroll
  for (int ti = 0; ti < 8; ++ti) {
    int tile = wid + 8 * ti;              // 0..63 : 8 w-tiles x 8 x-tiles
    int mw = tile >> 3, nx = tile & 7;
    v8f acc = {};
#pragma unroll
    for (int k0 = 0; k0 < C; k0 += 32) {
      v16bf a = fragA(sQp, LDP, mw * 16 + r16, k0, lane);   // A = qp [m=w][k=c]
      v16bf b = fragB(sKp, LDP, nx * 16 + r16, k0, lane);   // B = kp^T from [n=x][k=c]
      acc = wmma_bf16(a, b, acc);
    }
    const int x = nx * 16 + r16;
#pragma unroll
    for (int j = 0; j < 8; ++j)
      sAttn[(mw * 16 + halfA + j) * LDA + x] = (bf16)acc[j];
  }
  __syncthreads();

  // ---------------- Stage 3: out[w][c] = sum_x attn[w][x]*vp[x][c] ----------------
#pragma unroll
  for (int ti = 0; ti < 4; ++ti) {
    int tile = wid + 8 * ti;              // 0..31 : 8 w-tiles x 4 c-tiles
    int mw = tile >> 2, nc = tile & 3;
    v8f acc = {};
#pragma unroll
    for (int k0 = 0; k0 < W; k0 += 32) {
      v16bf a = fragA(sAttn, LDA, mw * 16 + r16, k0, lane);  // A = attn [m=w][k=x]
      v16bf b = fragB(sVpT, LDV, nc * 16 + r16, k0, lane);   // B = vp from [n=c][k=x]
      acc = wmma_bf16(a, b, acc);
    }
    const int c = nc * 16 + r16;
#pragma unroll
    for (int j = 0; j < 8; ++j)
      sOut[(mw * 16 + halfA + j) * LDO + c] = acc[j];
  }
  __syncthreads();

  // ---------------- Residual + NCHW writeback (coalesced float4) ----------------
  for (int t = tid; t < C * (W / 4); t += BDIM) {
    int c = t >> 5, w = (t & 31) << 2;
    const float4 f = *(const float4*)(qg + base + (size_t)c * HW + w);
    float4 o;
    o.x = f.x + sOut[(w + 0) * LDO + c];
    o.y = f.y + sOut[(w + 1) * LDO + c];
    o.z = f.z + sOut[(w + 2) * LDO + c];
    o.w = f.w + sOut[(w + 3) * LDO + c];
    *(float4*)(outg + base + (size_t)c * HW + w) = o;
  }
}

extern "C" void kernel_launch(void* const* d_in, const int* in_sizes, int n_in,
                              void* d_out, int out_size, void* d_ws, size_t ws_size,
                              hipStream_t stream) {
  const float* q  = (const float*)d_in[0];
  const float* k  = (const float*)d_in[1];
  const float* v  = (const float*)d_in[2];
  const float* wq = (const float*)d_in[3];
  const float* bq = (const float*)d_in[4];
  const float* wk = (const float*)d_in[5];
  const float* bk = (const float*)d_in[6];
  const float* wv = (const float*)d_in[7];
  const float* bv = (const float*)d_in[8];
  float* out = (float*)d_out;
  mha_w_attn_kernel<<<dim3(32 * 128), dim3(256), 0, stream>>>(
      q, k, v, wq, bq, wk, bk, wv, bv, out);
}
